// SelfAttention_1434519077571
// MI455X (gfx1250) — matrix operand order repaired and checked by reading, not compiled, not run on hardware
//
#include <hip/hip_runtime.h>
#include <hip/hip_bf16.h>
#include <math.h>

typedef __attribute__((ext_vector_type(16))) __bf16 v16bf;
typedef __attribute__((ext_vector_type(8)))  float  v8f;

#define BATCH   4
#define SEQ     2048
#define DEMB    1024
#define NHEADS  16
#define DHEAD   64
#define QKVDIM  3072
#define ROWS    (BATCH * SEQ)   // 8192

// A-fragment K index within a 32-wide chunk (contiguous order k = 16*hi + j, matching B fills)
__device__ __forceinline__ int a_kidx(int j, int hi) { return j + hi * 16; }

__device__ __forceinline__ v8f wmma_bf16(const v16bf& a, const v16bf& b, const v8f& c) {
    return __builtin_amdgcn_wmma_f32_16x16x32_bf16(false, a, false, b, (short)0, c, false, false);
}

// Async DMA: copy 16 bytes per lane from global to LDS (tracked by ASYNCcnt).
// VDST operand is the LDS byte address; low 32 bits of a flat LDS pointer are
// the LDS offset (ISA 10.2 aperture mapping).
__device__ __forceinline__ void async_copy_b128(void* lds_dst, const void* gsrc) {
    unsigned loff = (unsigned)(unsigned long long)lds_dst;
    asm volatile("global_load_async_to_lds_b128 %0, %1, off"
                 :: "v"(loff), "v"(gsrc) : "memory");
}
__device__ __forceinline__ void wait_async0() {
    asm volatile("s_wait_asynccnt 0x0" ::: "memory");
}

// ---------------------------------------------------------------------------
// One-time f32 -> bf16 conversion (n must be a multiple of 8)
// ---------------------------------------------------------------------------
__global__ void f32_to_bf16(const float* __restrict__ in, __bf16* __restrict__ out, int n) {
    int i = (blockIdx.x * blockDim.x + threadIdx.x) * 8;
    if (i >= n) return;
    float4 a = ((const float4*)(in + i))[0];
    float4 b = ((const float4*)(in + i))[1];
    out[i + 0] = (__bf16)a.x; out[i + 1] = (__bf16)a.y;
    out[i + 2] = (__bf16)a.z; out[i + 3] = (__bf16)a.w;
    out[i + 4] = (__bf16)b.x; out[i + 5] = (__bf16)b.y;
    out[i + 6] = (__bf16)b.z; out[i + 7] = (__bf16)b.w;
}

// ---------------------------------------------------------------------------
// C[M x N] = A[M x K] * B[N x K]^T + bias[N]   (bf16 in, f32 accumulate)
// Workgroup tile 128x128, 8 waves, wave tile 32x64. Async double-buffered
// K-tiles: DMA of tile k+1 overlaps WMMA on tile k.
// ---------------------------------------------------------------------------
template <int OUT_F32>
__global__ __launch_bounds__(256, 2)
void gemm_bias_wmma(const __bf16* __restrict__ A, const __bf16* __restrict__ B,
                    const float* __restrict__ bias, float* __restrict__ Cf,
                    __bf16* __restrict__ Cb, int M, int N, int K) {
    __shared__ __bf16 sA[2][128 * 32];
    __shared__ __bf16 sB[2][128 * 32];

    const int tid  = threadIdx.x;
    const int w    = tid >> 5;
    const int lane = tid & 31;
    const int l    = lane & 15;
    const int hi   = lane >> 4;

    const int m0 = blockIdx.y * 128;
    const int n0 = blockIdx.x * 128;
    const int wm = (w & 3) * 32;
    const int wn = (w >> 2) * 64;

    v8f acc[2][4];
#pragma unroll
    for (int jn = 0; jn < 4; ++jn) {
        float bv = bias[n0 + wn + jn * 16 + l];
#pragma unroll
        for (int i = 0; i < 2; ++i)
#pragma unroll
            for (int v = 0; v < 8; ++v)
                acc[i][jn][v] = bv;
    }

    // loader mapping: 256 threads x 32B cover one 128x32 bf16 tile (8 KB)
    const int lrow = tid >> 1;
    const int lseg = (tid & 1) * 16;   // element offset (16 bf16 = 32 B)
    const __bf16* Ag = A + (size_t)(m0 + lrow) * K + lseg;
    const __bf16* Bg = B + (size_t)(n0 + lrow) * K + lseg;
    __bf16* sAr = &sA[0][0] + lrow * 32 + lseg;
    __bf16* sBr = &sB[0][0] + lrow * 32 + lseg;
    const int bufstride = 128 * 32;

    // prologue: DMA tile 0 into buffer 0
    async_copy_b128(sAr,     Ag);
    async_copy_b128(sAr + 8, Ag + 8);
    async_copy_b128(sBr,     Bg);
    async_copy_b128(sBr + 8, Bg + 8);

    for (int k0 = 0; k0 < K; k0 += 32) {
        const int cur = (k0 >> 5) & 1;
        wait_async0();          // current tile landed in LDS
        __syncthreads();
        if (k0 + 32 < K) {      // overlap: DMA next tile while computing
            const int nxt = cur ^ 1;
            async_copy_b128(sAr + nxt * bufstride,     Ag + k0 + 32);
            async_copy_b128(sAr + nxt * bufstride + 8, Ag + k0 + 40);
            async_copy_b128(sBr + nxt * bufstride,     Bg + k0 + 32);
            async_copy_b128(sBr + nxt * bufstride + 8, Bg + k0 + 40);
        }

        v16bf afr[2];
#pragma unroll
        for (int i = 0; i < 2; ++i) {
            int mr = wm + i * 16 + l;
#pragma unroll
            for (int j = 0; j < 16; ++j)
                afr[i][j] = sA[cur][mr * 32 + a_kidx(j, hi)];
        }
#pragma unroll
        for (int jn = 0; jn < 4; ++jn) {
            v16bf bfr;
            int nr = wn + jn * 16 + l;
#pragma unroll
            for (int j = 0; j < 16; ++j)
                bfr[j] = sB[cur][nr * 32 + j + hi * 16];
#pragma unroll
            for (int i = 0; i < 2; ++i)
                acc[i][jn] = wmma_bf16(afr[i], bfr, acc[i][jn]);
        }
        __syncthreads();
    }

#pragma unroll
    for (int i = 0; i < 2; ++i)
#pragma unroll
        for (int jn = 0; jn < 4; ++jn)
#pragma unroll
            for (int v = 0; v < 8; ++v) {
                int m = m0 + wm + i * 16 + hi * 8 + v;
                int n = n0 + wn + jn * 16 + l;
                if (OUT_F32) Cf[(size_t)m * N + n] = acc[i][jn][v];
                else         Cb[(size_t)m * N + n] = (__bf16)acc[i][jn][v];
            }
}

// ---------------------------------------------------------------------------
// Flash attention over bf16 qkv[b, s, 3*DEMB]. One block = (b, h, 128 q rows);
// 8 waves x 16 q rows; K-tiles of 64 keys. K staged via async DMA (row-major
// is already B-fragment friendly), V staged transposed. Output bf16
// attn[b, s, h*DHEAD + d].
// ---------------------------------------------------------------------------
__global__ __launch_bounds__(256, 2)
void attn_kernel(const __bf16* __restrict__ qkv, __bf16* __restrict__ attn) {
    __shared__ __bf16 sK [64 * 64];        // [key][dh]
    __shared__ __bf16 sVt[64 * 64];        // [dh][key] (transposed)
    __shared__ __bf16 sP [8 * 16 * 64];    // per-wave P scratch

    const int tid  = threadIdx.x;
    const int w    = tid >> 5;
    const int lane = tid & 31;
    const int l    = lane & 15;
    const int hi   = lane >> 4;

    const int bh = blockIdx.y;
    const int b  = bh >> 4;
    const int h  = bh & 15;
    const int q0 = blockIdx.x * 128;

    const __bf16* base = qkv + (size_t)b * SEQ * QKVDIM;
    const int qoff = h * DHEAD;
    const int koff = DEMB + h * DHEAD;
    const int voff = 2 * DEMB + h * DHEAD;
    const float scale = 0.125f; // 1/sqrt(64)

    // Q fragments for this wave's 16 rows (2 chunks of K=32)
    v16bf aQ[2];
    {
        const __bf16* qrow = base + (size_t)(q0 + w * 16 + l) * QKVDIM + qoff;
#pragma unroll
        for (int c = 0; c < 2; ++c)
#pragma unroll
            for (int j = 0; j < 16; ++j)
                aQ[c][j] = qrow[c * 32 + a_kidx(j, hi)];
    }

    float mrow[8], lsum[8];
    v8f o[4];
#pragma unroll
    for (int v = 0; v < 8; ++v) { mrow[v] = -__builtin_inff(); lsum[v] = 0.f; }
#pragma unroll
    for (int jn = 0; jn < 4; ++jn)
#pragma unroll
        for (int v = 0; v < 8; ++v) o[jn][v] = 0.f;

    const int qhi = q0 + w * 16 + 15;      // wave-uniform causal bound
    const int kr  = tid >> 1;              // K loader: key row (0..127 -> but 64 rows, 2 segs)
    const int ksg = (tid & 1) * 16;        // K loader: 32B segment  (wait: 64 rows x 2 segs = 128... )
    const int vr  = tid >> 2;              // V loader: key row
    const int vcs = (tid & 3) * 16;        // V loader: dh col segment

    for (int k0 = 0; k0 < q0 + 128; k0 += 64) {
        // ---- K tile: straight async DMA (64x64 bf16 = 8 KB, 32 B/thread) ----
        {
            const __bf16* kg = base + (size_t)(k0 + (kr & 63)) * QKVDIM + koff + ksg;
            __bf16* kd = sK + (kr & 63) * 64 + ksg;
            async_copy_b128(kd,     kg);
            async_copy_b128(kd + 8, kg + 8);
        }
        // ---- V tile: stage transposed via VGPRs ----
        {
            const __bf16* vg = base + (size_t)(k0 + vr) * QKVDIM + voff + vcs;
#pragma unroll
            for (int j = 0; j < 16; ++j)
                sVt[(vcs + j) * 64 + vr] = vg[j];
        }
        wait_async0();
        __syncthreads();

        if (k0 <= qhi) {                   // wave-uniform branch: EXEC stays full
            // ---- S = Q K^T (16x64) ----
            v8f sc[4];
#pragma unroll
            for (int jn = 0; jn < 4; ++jn)
#pragma unroll
                for (int v = 0; v < 8; ++v) sc[jn][v] = 0.f;
#pragma unroll
            for (int c = 0; c < 2; ++c)
#pragma unroll
                for (int jn = 0; jn < 4; ++jn) {
                    v16bf bk;
#pragma unroll
                    for (int j = 0; j < 16; ++j)
                        bk[j] = sK[(jn * 16 + l) * 64 + c * 32 + hi * 16 + j];
                    sc[jn] = wmma_bf16(aQ[c], bk, sc[jn]);
                }

            // ---- scale, causal mask, online softmax ----
            float corr[8];
#pragma unroll
            for (int v = 0; v < 8; ++v) {
                int qi = q0 + w * 16 + hi * 8 + v;
                float rmax = -__builtin_inff();
#pragma unroll
                for (int jn = 0; jn < 4; ++jn) {
                    int kj = k0 + jn * 16 + l;
                    float x = sc[jn][v] * scale;
                    if (kj > qi) x = -__builtin_inff();
                    sc[jn][v] = x;
                    rmax = fmaxf(rmax, x);
                }
#pragma unroll
                for (int off = 1; off < 16; off <<= 1)
                    rmax = fmaxf(rmax, __shfl_xor(rmax, off, 32));
                float mnew = fmaxf(mrow[v], rmax);
                corr[v] = __expf(mrow[v] - mnew);
                float rsum = 0.f;
#pragma unroll
                for (int jn = 0; jn < 4; ++jn) {
                    float p = __expf(sc[jn][v] - mnew);
                    sc[jn][v] = p;
                    rsum += p;
                }
#pragma unroll
                for (int off = 1; off < 16; off <<= 1)
                    rsum += __shfl_xor(rsum, off, 32);
                lsum[v] = lsum[v] * corr[v] + rsum;
                mrow[v] = mnew;
            }
#pragma unroll
            for (int jn = 0; jn < 4; ++jn)
#pragma unroll
                for (int v = 0; v < 8; ++v) o[jn][v] *= corr[v];

            // ---- P to per-wave LDS scratch (C-layout -> A-layout shuffle) ----
            __bf16* pw = sP + w * (16 * 64);
#pragma unroll
            for (int jn = 0; jn < 4; ++jn)
#pragma unroll
                for (int v = 0; v < 8; ++v)
                    pw[(hi * 8 + v) * 64 + jn * 16 + l] = (__bf16)sc[jn][v];
            asm volatile("s_wait_dscnt 0x0" ::: "memory"); // intra-wave LDS RAW

            // ---- O += P V ----
#pragma unroll
            for (int c = 0; c < 2; ++c) {
                v16bf aP;
#pragma unroll
                for (int j = 0; j < 16; ++j)
                    aP[j] = pw[l * 64 + c * 32 + a_kidx(j, hi)];
#pragma unroll
                for (int jn = 0; jn < 4; ++jn) {
                    v16bf bv;
#pragma unroll
                    for (int j = 0; j < 16; ++j)
                        bv[j] = sVt[(jn * 16 + l) * 64 + c * 32 + hi * 16 + j];
                    o[jn] = wmma_bf16(aP, bv, o[jn]);
                }
            }
        }
        __syncthreads();
    }

    // ---- epilogue: normalize and store bf16 [b, s, h*64 + d] ----
#pragma unroll
    for (int jn = 0; jn < 4; ++jn)
#pragma unroll
        for (int v = 0; v < 8; ++v) {
            int qi = q0 + w * 16 + hi * 8 + v;
            float val = o[jn][v] / lsum[v];
            attn[((size_t)b * SEQ + qi) * DEMB + h * DHEAD + jn * 16 + l] = (__bf16)val;
        }
}

extern "C" void kernel_launch(void* const* d_in, const int* in_sizes, int n_in,
                              void* d_out, int out_size, void* d_ws, size_t ws_size,
                              hipStream_t stream) {
    const float* x     = (const float*)d_in[0];   // [4,2048,1024]
    const float* W_in  = (const float*)d_in[1];   // [3072,1024]
    const float* b_in  = (const float*)d_in[2];   // [3072]
    const float* W_out = (const float*)d_in[3];   // [1024,1024]
    const float* b_out = (const float*)d_in[4];   // [1024]
    float* out = (float*)d_out;                   // [4,2048,1024]

    // workspace layout (bf16): xb | W_inb | W_outb | qkvb | attnb  (~88 MB)
    __bf16* xb    = (__bf16*)d_ws;
    __bf16* wib   = xb   + (size_t)ROWS * DEMB;
    __bf16* wob   = wib  + (size_t)QKVDIM * DEMB;
    __bf16* qkvb  = wob  + (size_t)DEMB * DEMB;
    __bf16* attnb = qkvb + (size_t)ROWS * QKVDIM;

    dim3 block(256);

    // 0) one-time bf16 conversions
    f32_to_bf16<<<(ROWS * DEMB / 8 + 255) / 256, block, 0, stream>>>(x, xb, ROWS * DEMB);
    f32_to_bf16<<<(QKVDIM * DEMB / 8 + 255) / 256, block, 0, stream>>>(W_in, wib, QKVDIM * DEMB);
    f32_to_bf16<<<(DEMB * DEMB / 8 + 255) / 256, block, 0, stream>>>(W_out, wob, DEMB * DEMB);

    // 1) qkv = x @ W_in^T + b_in   (bf16 out)
    gemm_bias_wmma<0><<<dim3(QKVDIM / 128, ROWS / 128), block, 0, stream>>>(
        xb, wib, b_in, nullptr, qkvb, ROWS, QKVDIM, DEMB);

    // 2) causal multi-head flash attention (bf16 out)
    attn_kernel<<<dim3(SEQ / 128, BATCH * NHEADS), block, 0, stream>>>(qkvb, attnb);

    // 3) out = attn @ W_out^T + b_out   (f32 out)
    gemm_bias_wmma<1><<<dim3(DEMB / 128, ROWS / 128), block, 0, stream>>>(
        attnb, wob, b_out, out, nullptr, ROWS, DEMB, DEMB);
}
